// TransitionPathE3NNLayer_82454782148790
// MI455X (gfx1250) — matrix operand order repaired and checked by reading, not compiled, not run on hardware
//
#include <hip/hip_runtime.h>
#include <hip/hip_bf16.h>
#include <math.h>

typedef __attribute__((ext_vector_type(16))) _Float16 v16h;
typedef __attribute__((ext_vector_type(8)))  _Float16 v8h;
typedef __attribute__((ext_vector_type(8)))  float    v8f;

#define N_NODES 4096
#define N_EDGES 65536
#define MUL0 32
#define MUL1 16
#define F_DIM 80
#define N_EF 37
#define EF_PAD 40
#define CTX_D 101
#define K_PAD 128
#define W_TP 2304
#define D_CUT 5.0f
#define N_RBF 10

__device__ __forceinline__ float gelu_exact(float x) {
  return 0.5f * x * (1.0f + erff(x * 0.70710678118654752f));
}
__device__ __forceinline__ float sigmoidf_(float x) { return 1.0f / (1.0f + expf(-x)); }

// ---- WMMA fragment loaders (wave32, contiguous k = 16*hi + s order in A and B) ----
// A 16x32 f16 from row-major [16][ld] LDS. Per lane (m=L&15, hi=L>>4) the 16
// halves are two contiguous 16B runs: k = hi*16+0..7 (slots 0-7) and
// k = hi*16+8..15 (slots 8-15)  ->  2x ds_load_b128.
union AFrag { v16h v; v8h h[2]; };
__device__ __forceinline__ v16h load_a_frag(const _Float16* __restrict__ buf, int ld, int k0) {
  const int lane = threadIdx.x & 31;
  const int m = lane & 15, hi = lane >> 4;
  const _Float16* p = buf + m * ld + k0 + hi * 16;
  AFrag a;
  a.h[0] = *(const v8h*)p;          // k = hi*16 + 0..7
  a.h[1] = *(const v8h*)(p + 8);    // k = hi*16 + 8..15
  return a.v;
}
// B 32x16 f16 from fragment-contiguous repacked weights: tile t stores
// 32 lanes x 16 halves contiguously -> one aligned v16h (2x global_load_b128).
__device__ __forceinline__ v16h load_b_frag(const _Float16* __restrict__ w, int tile) {
  const int lane = threadIdx.x & 31;
  return *(const v16h*)(w + ((size_t)(tile * 32 + lane)) * 16);
}

// ---------------- utility kernels ----------------
__global__ void zero_kernel(float* __restrict__ p, int n) {
  for (int i = blockIdx.x * blockDim.x + threadIdx.x; i < n; i += gridDim.x * blockDim.x)
    p[i] = 0.0f;
}

// Repack W[K][N] (f32) -> f16 fragment-contiguous tiles:
// out[((kt*ntiles+nt)*32 + lane)*16 + s] = W[kt*32 + s + 16*(lane>>4)][nt*16 + (lane&15)]
__device__ __forceinline__ void pack_frag(const float* __restrict__ W, int K, int N, int idx,
                                          _Float16* __restrict__ out) {
  const int s = idx & 15;
  const int lane = (idx >> 4) & 31;
  const int tile = idx >> 9;
  const int ntiles = N >> 4;
  const int kt = tile / ntiles, nt = tile - kt * ntiles;
  const int k = kt * 32 + s + ((lane >> 4) << 4);
  const int n = nt * 16 + (lane & 15);
  out[idx] = (_Float16)((k < K) ? W[k * N + n] : 0.0f);
}

__global__ void prep_weights_kernel(const float* __restrict__ rW1, const float* __restrict__ rW2,
                                    const float* __restrict__ rW3, const float* __restrict__ ecW1,
                                    const float* __restrict__ ecW2,
                                    _Float16* __restrict__ rW1h, _Float16* __restrict__ rW2h,
                                    _Float16* __restrict__ rW3h, _Float16* __restrict__ ecW1h,
                                    _Float16* __restrict__ ecW2h) {
  const int total = 8192 + 8192 + 4096 + 4096 + 64 * W_TP;
  for (int i = blockIdx.x * blockDim.x + threadIdx.x; i < total; i += gridDim.x * blockDim.x) {
    int idx = i;
    if (idx < 8192) { pack_frag(rW1, CTX_D, 64, idx, rW1h); continue; }   // Kp=128
    idx -= 8192;
    if (idx < 8192) { pack_frag(ecW1, CTX_D, 64, idx, ecW1h); continue; } // Kp=128
    idx -= 8192;
    if (idx < 4096) { pack_frag(rW2, 64, 64, idx, rW2h); continue; }
    idx -= 4096;
    if (idx < 4096) { pack_frag(ecW2, 64, 64, idx, ecW2h); continue; }
    idx -= 4096;
    pack_frag(rW3, 64, W_TP, idx, rW3h);
  }
}

// node_scalars = f[:, :32] @ W_readout
__global__ void node_scalars_kernel(const float* __restrict__ f, const float* __restrict__ Wr,
                                    float* __restrict__ ns) {
  int idx = blockIdx.x * blockDim.x + threadIdx.x;
  if (idx >= N_NODES * 32) return;
  int n = idx >> 5, o = idx & 31;
  float a = 0.0f;
  for (int j = 0; j < 32; ++j) a += f[n * F_DIM + j] * Wr[j * 32 + o];
  ns[idx] = a;
}

// ---------------- edge message kernel (WMMA MLP + fused tensor product) ----------------
__global__ __launch_bounds__(256) void edge_msg_kernel(
    const float* __restrict__ f, const float* __restrict__ x,
    const float* __restrict__ xA, const float* __restrict__ xB,
    const int* __restrict__ src, const int* __restrict__ dst,
    const int* __restrict__ bondA, const int* __restrict__ bondB,
    const float* __restrict__ ns,
    const _Float16* __restrict__ rW1h, const float* __restrict__ rb1,
    const _Float16* __restrict__ rW2h, const float* __restrict__ rb2,
    const _Float16* __restrict__ rW3h, const float* __restrict__ rb3,
    float* __restrict__ efeat, float* __restrict__ agg_s, float* __restrict__ agg_v) {
  __shared__ __align__(32) _Float16 sCtx[16 * K_PAD];
  __shared__ __align__(32) _Float16 sH1[16 * 64];
  __shared__ __align__(32) _Float16 sH2[16 * 64];
  __shared__ __align__(16) float    sW[16 * 128];
  __shared__ float sFeat[16 * EF_PAD];
  __shared__ float sSS[16 * 32];
  __shared__ float sVV[16 * 48];
  __shared__ float sSH[16 * 4];
  __shared__ float sVD[16 * 16];

  const int tid = threadIdx.x;
  const int wv  = tid >> 5;
  const int e0  = blockIdx.x * 16;

  // phase 0a: per-edge scalar features (one thread per edge)
  if (tid < 16) {
    const int e = e0 + tid;
    const int is = src[e], id = dst[e];
    float ev[3], evA[3], evB[3];
#pragma unroll
    for (int c = 0; c < 3; ++c) {
      ev[c]  = x[id * 3 + c]  - x[is * 3 + c];
      evA[c] = xA[id * 3 + c] - xA[is * 3 + c];
      evB[c] = xB[id * 3 + c] - xB[is * 3 + c];
    }
    float d2  = fmaxf(ev[0] * ev[0] + ev[1] * ev[1] + ev[2] * ev[2], 1e-8f);
    float dA2 = fmaxf(evA[0] * evA[0] + evA[1] * evA[1] + evA[2] * evA[2], 1e-8f);
    float dB2 = fmaxf(evB[0] * evB[0] + evB[1] * evB[1] + evB[2] * evB[2], 1e-8f);
    float draw = sqrtf(d2), dAraw = sqrtf(dA2), dBraw = sqrtf(dB2);
    float inv = 1.0f / draw;
    float d = draw / D_CUT, dA = dAraw / D_CUT, dB = dBraw / D_CUT;
    float ft[N_EF];
    ft[0] = (float)bondA[e]; ft[1] = (float)bondB[e];
    ft[2] = d; ft[3] = d * d; ft[4] = dA; ft[5] = dB; ft[6] = dA - dB;
    const float rbw = D_CUT / (float)(N_RBF - 1);
    const float irbw = 1.0f / rbw;
#pragma unroll
    for (int k = 0; k < N_RBF; ++k) {
      float ck = rbw * (float)k;
      float t0 = (draw - ck) * irbw, t1 = (dAraw - ck) * irbw, t2 = (dBraw - ck) * irbw;
      ft[7 + k]  = expf(-0.5f * t0 * t0);
      ft[17 + k] = expf(-0.5f * t1 * t1);
      ft[27 + k] = expf(-0.5f * t2 * t2);
    }
#pragma unroll
    for (int j = 0; j < N_EF; ++j) {
      sFeat[tid * EF_PAD + j] = ft[j];
      efeat[e * EF_PAD + j] = ft[j];
    }
    const float s3 = 1.7320508075688772f;
#pragma unroll
    for (int c = 0; c < 3; ++c) sSH[tid * 4 + c] = s3 * ev[c] * inv;  // sh1
    sSH[tid * 4 + 3] = 0.0f;
  }
  __syncthreads();

  // phase 0b: gather source-node scalar/vector features
  for (int idx = tid; idx < 16 * 32; idx += 256) {
    int e = idx >> 5, i = idx & 31;
    sSS[idx] = f[src[e0 + e] * F_DIM + i];
  }
  for (int idx = tid; idx < 16 * 48; idx += 256) {
    int e = idx / 48, q = idx % 48;
    sVV[idx] = f[src[e0 + e] * F_DIM + MUL0 + q];
  }
  __syncthreads();
  for (int idx = tid; idx < 16 * 16; idx += 256) {
    int e = idx >> 4, i = idx & 15;
    sVD[idx] = sVV[e * 48 + i * 3 + 0] * sSH[e * 4 + 0] +
               sVV[e * 48 + i * 3 + 1] * sSH[e * 4 + 1] +
               sVV[e * 48 + i * 3 + 2] * sSH[e * 4 + 2];
  }
  // ctx assembly (f16, K padded to 128)
  for (int idx = tid; idx < 16 * K_PAD; idx += 256) {
    int e = idx >> 7, c = idx & 127;
    float v;
    if (c < N_EF)            v = sFeat[e * EF_PAD + c];
    else if (c < N_EF + 32)  v = ns[src[e0 + e] * 32 + (c - N_EF)];
    else if (c < CTX_D)      v = ns[dst[e0 + e] * 32 + (c - N_EF - 32)];
    else                     v = 0.0f;
    sCtx[idx] = (_Float16)v;
  }
  __syncthreads();

  // layer 1: [16x128] @ [128x64]  (waves 0..3, one 16-wide N tile each; ntiles=4)
  if (wv < 4) {
    v8f acc = {};
#pragma unroll
    for (int k0 = 0; k0 < K_PAD; k0 += 32) {
      v16h a = load_a_frag(sCtx, K_PAD, k0);
      v16h b = load_b_frag(rW1h, (k0 >> 5) * 4 + wv);
      acc = __builtin_amdgcn_wmma_f32_16x16x32_f16(false, a, false, b, (short)0, acc, false, false);
    }
    const int lane = tid & 31, n = lane & 15, hi = lane >> 4;
    const int col = wv * 16 + n;
    const float bias = rb1[col];
#pragma unroll
    for (int r = 0; r < 8; ++r) {
      int m = r + 8 * hi;
      sH1[m * 64 + col] = (_Float16)gelu_exact(acc[r] + bias);
    }
  }
  __syncthreads();

  // layer 2: [16x64] @ [64x64]  (ntiles=4)
  if (wv < 4) {
    v8f acc = {};
#pragma unroll
    for (int k0 = 0; k0 < 64; k0 += 32) {
      v16h a = load_a_frag(sH1, 64, k0);
      v16h b = load_b_frag(rW2h, (k0 >> 5) * 4 + wv);
      acc = __builtin_amdgcn_wmma_f32_16x16x32_f16(false, a, false, b, (short)0, acc, false, false);
    }
    const int lane = tid & 31, n = lane & 15, hi = lane >> 4;
    const int col = wv * 16 + n;
    const float bias = rb2[col];
#pragma unroll
    for (int r = 0; r < 8; ++r) {
      int m = r + 8 * hi;
      sH2[m * 64 + col] = (_Float16)gelu_exact(acc[r] + bias);
    }
  }
  __syncthreads();

  // layer 3 fused with tensor-product consumption:
  // w[16 x 2304] computed in 128-col chunks (8 waves x 1 WMMA tile), consumed immediately.
  float ms0 = 0.0f, ms1 = 0.0f, mv0 = 0.0f, mv1 = 0.0f, mv2 = 0.0f;
  const int ce = tid >> 4;  // edge within tile
  const int t  = tid & 15;  // owned output index
  const float shx = sSH[ce * 4 + 0], shy = sSH[ce * 4 + 1], shz = sSH[ce * 4 + 2];
  const float S32 = 0.17677669529663687f;   // 1/sqrt(32)
  const float S48 = 0.14433756729740643f;   // 1/sqrt(48)
  const float S16 = 0.25f;                  // 1/sqrt(16)
  const int NT3 = W_TP / 16;                // 144 N tiles in rW3h

  for (int ch = 0; ch < W_TP / 128; ++ch) {
    // prefetch next chunk's B tiles for this wave (L2-resident stream)
    if (ch + 1 < W_TP / 128) {
      const int nt = (ch + 1) * 8 + wv;
      __builtin_prefetch((const char*)(rW3h + ((size_t)(nt * 32 + (tid & 31))) * 16), 0, 1);
      __builtin_prefetch((const char*)(rW3h + ((size_t)((NT3 + nt) * 32 + (tid & 31))) * 16), 0, 1);
    }
    v8f acc = {};
#pragma unroll
    for (int k0 = 0; k0 < 64; k0 += 32) {
      v16h a = load_a_frag(sH2, 64, k0);
      v16h b = load_b_frag(rW3h, (k0 >> 5) * NT3 + ch * 8 + wv);
      acc = __builtin_amdgcn_wmma_f32_16x16x32_f16(false, a, false, b, (short)0, acc, false, false);
    }
    {
      const int lane = tid & 31, n = lane & 15, hi = lane >> 4;
      const int cl = wv * 16 + n;
      const float bias = rb3[ch * 128 + cl];
#pragma unroll
      for (int r = 0; r < 8; ++r) {
        int m = r + 8 * hi;
        sW[m * 128 + cl] = acc[r] + bias;
      }
    }
    __syncthreads();
#pragma unroll
    for (int L8 = 0; L8 < 8; ++L8) {
      const int L = L8 * 16 + t;
      const int j = ch * 128 + L;      // global column; j mod 16 == t
      const float wval = sW[ce * 128 + L];
      if (j < 1024) {                  // w00[i,jj]: msg_s += ss[i]*w / sqrt32
        int i = j >> 5;
        float c = sSS[ce * 32 + i] * wval * S32;
        if (j & 16) ms1 += c; else ms0 += c;
      } else if (j < 1536) {           // w01[i,jj]: msg_v += ss[i]*sh1*w / sqrt32
        int i = (j - 1024) >> 4;
        float c = sSS[ce * 32 + i] * wval * S32;
        mv0 += c * shx; mv1 += c * shy; mv2 += c * shz;
      } else if (j < 1792) {           // w10[i,jj]: msg_v += vv[i]*w / sqrt16
        int i = (j - 1536) >> 4;
        float c = wval * S16;
        mv0 += c * sVV[ce * 48 + i * 3 + 0];
        mv1 += c * sVV[ce * 48 + i * 3 + 1];
        mv2 += c * sVV[ce * 48 + i * 3 + 2];
      } else {                         // w11[i,jj]: msg_s += (vv[i].sh1)*w / sqrt48
        int i = (j - 1792) >> 5;
        float c = sVD[ce * 16 + i] * wval * S48;
        if (j & 16) ms1 += c; else ms0 += c;
      }
    }
    __syncthreads();
  }

  // scatter into segment sums
  {
    const int de = dst[e0 + ce];
    atomicAdd(&agg_s[de * 32 + t], ms0);
    atomicAdd(&agg_s[de * 32 + t + 16], ms1);
    atomicAdd(&agg_v[de * 48 + t * 3 + 0], mv0);
    atomicAdd(&agg_v[de * 48 + t * 3 + 1], mv1);
    atomicAdd(&agg_v[de * 48 + t * 3 + 2], mv2);
  }
}

// ---------------- node update (gating, f_new, delta_x, readout, cg-MLP) ----------------
__global__ void node_update_kernel(
    const float* __restrict__ f, const float* __restrict__ agg_s, const float* __restrict__ agg_v,
    const float* __restrict__ W_si_s, const float* __restrict__ W_si_v,
    const float* __restrict__ W_pg_s, const float* __restrict__ W_pg_v,
    const float* __restrict__ W_readout, const float* __restrict__ W_coord,
    const float* __restrict__ cgW1, const float* __restrict__ cgb1,
    const float* __restrict__ cgW2, const float* __restrict__ cgb2,
    const float* __restrict__ cgW3, const float* __restrict__ cgb3,
    float* __restrict__ fout, float* __restrict__ scal,
    float* __restrict__ dxv, float* __restrict__ g) {
  const int n = blockIdx.x * blockDim.x + threadIdx.x;
  if (n >= N_NODES) return;
  float sis[32];
  for (int o = 0; o < 32; ++o) {
    float a = 0.0f;
    for (int i = 0; i < 32; ++i) a += agg_s[n * 32 + i] * W_si_s[i * 32 + o];
    sis[o] = a;
  }
  float siv[48];
  for (int o = 0; o < 16; ++o)
    for (int c = 0; c < 3; ++c) {
      float a = 0.0f;
      for (int i = 0; i < 16; ++i) a += agg_v[n * 48 + i * 3 + c] * W_si_v[i * 16 + o];
      siv[o * 3 + c] = a;
    }
  float pgs[48];
  for (int o = 0; o < 48; ++o) {
    float a = 0.0f;
    for (int i = 0; i < 32; ++i) a += sis[i] * W_pg_s[i * 48 + o];
    pgs[o] = a;
  }
  float pgv[48];
  for (int o = 0; o < 16; ++o)
    for (int c = 0; c < 3; ++c) {
      float a = 0.0f;
      for (int i = 0; i < 16; ++i) a += siv[i * 3 + c] * W_pg_v[i * 16 + o];
      pgv[o * 3 + c] = a;
    }
  float sfn[32];
  for (int j = 0; j < 32; ++j) {
    float u = pgs[j] * sigmoidf_(pgs[j]);  // silu
    float v = f[n * F_DIM + j] + 0.2f * u;
    fout[n * F_DIM + j] = v;
    sfn[j] = v;
  }
  float vfn[48];
  for (int q = 0; q < 48; ++q) {
    float u = sigmoidf_(pgs[32 + q / 3]) * pgv[q];
    float v = f[n * F_DIM + 32 + q] + 0.2f * u;
    fout[n * F_DIM + 32 + q] = v;
    vfn[q] = v;
  }
  for (int c = 0; c < 3; ++c) {
    float a = 0.0f;
    for (int i = 0; i < 16; ++i) a += vfn[i * 3 + c] * W_coord[i];
    dxv[n * 3 + c] = a;
  }
  float sc[32];
  for (int o = 0; o < 32; ++o) {
    float a = 0.0f;
    for (int j = 0; j < 32; ++j) a += sfn[j] * W_readout[j * 32 + o];
    sc[o] = a;
    scal[n * 32 + o] = a;
  }
  float h1[64];
  for (int o = 0; o < 64; ++o) {
    float a = cgb1[o];
    for (int j = 0; j < 32; ++j) a += sc[j] * cgW1[j * 64 + o];
    h1[o] = gelu_exact(a);
  }
  float h2[64];
  for (int o = 0; o < 64; ++o) {
    float a = cgb2[o];
    for (int j = 0; j < 64; ++j) a += h1[j] * cgW2[j * 64 + o];
    h2[o] = gelu_exact(a);
  }
  for (int o = 0; o < 4; ++o) {
    float a = cgb3[o];
    for (int j = 0; j < 64; ++j) a += h2[j] * cgW3[j * 4 + o];
    g[n * 4 + o] = a;
  }
}

// ---------------- edge coordination kernel (second MLP via WMMA) ----------------
__global__ __launch_bounds__(256) void edge_coord_kernel(
    const float* __restrict__ x, const int* __restrict__ src, const int* __restrict__ dst,
    const float* __restrict__ efeat, const float* __restrict__ scal,
    const _Float16* __restrict__ ecW1h, const float* __restrict__ ecb1,
    const _Float16* __restrict__ ecW2h, const float* __restrict__ ecb2,
    const float* __restrict__ ecW3, const float* __restrict__ ecb3,
    float* __restrict__ nupd) {
  __shared__ __align__(32) _Float16 sCtx[16 * K_PAD];
  __shared__ __align__(32) _Float16 sH1[16 * 64];
  __shared__ __align__(32) _Float16 sH2[16 * 64];
  const int tid = threadIdx.x;
  const int wv  = tid >> 5;
  const int e0  = blockIdx.x * 16;

  for (int idx = tid; idx < 16 * K_PAD; idx += 256) {
    int e = idx >> 7, c = idx & 127;
    float v;
    if (c < N_EF)            v = efeat[(e0 + e) * EF_PAD + c];
    else if (c < N_EF + 32)  v = scal[src[e0 + e] * 32 + (c - N_EF)];
    else if (c < CTX_D)      v = scal[dst[e0 + e] * 32 + (c - N_EF - 32)];
    else                     v = 0.0f;
    sCtx[idx] = (_Float16)v;
  }
  __syncthreads();
  if (wv < 4) {
    v8f acc = {};
#pragma unroll
    for (int k0 = 0; k0 < K_PAD; k0 += 32) {
      v16h a = load_a_frag(sCtx, K_PAD, k0);
      v16h b = load_b_frag(ecW1h, (k0 >> 5) * 4 + wv);
      acc = __builtin_amdgcn_wmma_f32_16x16x32_f16(false, a, false, b, (short)0, acc, false, false);
    }
    const int lane = tid & 31, n = lane & 15, hi = lane >> 4;
    const int col = wv * 16 + n;
    const float bias = ecb1[col];
#pragma unroll
    for (int r = 0; r < 8; ++r) {
      int m = r + 8 * hi;
      sH1[m * 64 + col] = (_Float16)gelu_exact(acc[r] + bias);
    }
  }
  __syncthreads();
  if (wv < 4) {
    v8f acc = {};
#pragma unroll
    for (int k0 = 0; k0 < 64; k0 += 32) {
      v16h a = load_a_frag(sH1, 64, k0);
      v16h b = load_b_frag(ecW2h, (k0 >> 5) * 4 + wv);
      acc = __builtin_amdgcn_wmma_f32_16x16x32_f16(false, a, false, b, (short)0, acc, false, false);
    }
    const int lane = tid & 31, n = lane & 15, hi = lane >> 4;
    const int col = wv * 16 + n;
    const float bias = ecb2[col];
#pragma unroll
    for (int r = 0; r < 8; ++r) {
      int m = r + 8 * hi;
      sH2[m * 64 + col] = (_Float16)gelu_exact(acc[r] + bias);
    }
  }
  __syncthreads();
  if (tid < 16) {
    const int e = e0 + tid;
    float step = ecb3[0];
    for (int q = 0; q < 64; ++q) step += (float)sH2[tid * 64 + q] * ecW3[q];
    const int is = src[e], id = dst[e];
#pragma unroll
    for (int c = 0; c < 3; ++c) {
      float dx = x[is * 3 + c] - x[id * 3 + c];  // dx_to_src = -ev
      atomicAdd(&nupd[id * 3 + c], step * dx);
    }
  }
}

// ---------------- final coordinate blend ----------------
__global__ void coord_final_kernel(const float* __restrict__ x, const float* __restrict__ xA,
                                   const float* __restrict__ xB, const float* __restrict__ s,
                                   const float* __restrict__ g, const float* __restrict__ dxv,
                                   const float* __restrict__ nupd, float* __restrict__ xout) {
  const int n = blockIdx.x * blockDim.x + threadIdx.x;
  if (n >= N_NODES) return;
  const float g0 = g[n * 4 + 0], g1 = g[n * 4 + 1], g2 = g[n * 4 + 2], g3 = g[n * 4 + 3];
  const float sn = s[n];
#pragma unroll
  for (int c = 0; c < 3; ++c) {
    const float xc = x[n * 3 + c];
    xout[n * 3 + c] = xc + g0 * dxv[n * 3 + c] + g1 * nupd[n * 3 + c] +
                      g2 * (1.0f - sn) * (xA[n * 3 + c] - xc) +
                      g3 * sn * (xB[n * 3 + c] - xc);
  }
}

extern "C" void kernel_launch(void* const* d_in, const int* in_sizes, int n_in,
                              void* d_out, int out_size, void* d_ws, size_t ws_size,
                              hipStream_t stream) {
  const float* f   = (const float*)d_in[0];
  const float* x   = (const float*)d_in[1];
  const float* xA  = (const float*)d_in[2];
  const float* xB  = (const float*)d_in[3];
  const float* s   = (const float*)d_in[4];
  const int* src   = (const int*)d_in[5];
  const int* dst   = (const int*)d_in[6];
  const int* bondA = (const int*)d_in[7];
  const int* bondB = (const int*)d_in[8];
  const float* W_readout = (const float*)d_in[9];
  const float* rW1 = (const float*)d_in[10]; const float* rb1 = (const float*)d_in[11];
  const float* rW2 = (const float*)d_in[12]; const float* rb2 = (const float*)d_in[13];
  const float* rW3 = (const float*)d_in[14]; const float* rb3 = (const float*)d_in[15];
  const float* W_si_s = (const float*)d_in[16];
  const float* W_si_v = (const float*)d_in[17];
  const float* W_pg_s = (const float*)d_in[18];
  const float* W_pg_v = (const float*)d_in[19];
  const float* cgW1 = (const float*)d_in[20]; const float* cgb1 = (const float*)d_in[21];
  const float* cgW2 = (const float*)d_in[22]; const float* cgb2 = (const float*)d_in[23];
  const float* cgW3 = (const float*)d_in[24]; const float* cgb3 = (const float*)d_in[25];
  const float* W_coord = (const float*)d_in[26];
  const float* ecW1 = (const float*)d_in[27]; const float* ecb1 = (const float*)d_in[28];
  const float* ecW2 = (const float*)d_in[29]; const float* ecb2 = (const float*)d_in[30];
  const float* ecW3 = (const float*)d_in[31]; const float* ecb3 = (const float*)d_in[32];

  char* ws = (char*)d_ws;
  float* ns    = (float*)(ws + 0);                     // 4096*32 f32
  float* scal  = (float*)(ws + 524288);                // 4096*32 f32
  float* agg_s = (float*)(ws + 1048576);               // 4096*32 f32
  float* agg_v = (float*)(ws + 1572864);               // 4096*48 f32
  float* nupd  = (float*)(ws + 2359296);               // 4096*3  f32
  float* dxv   = (float*)(ws + 2408448);               // 4096*3  f32
  float* g     = (float*)(ws + 2457600);               // 4096*4  f32
  float* efeat = (float*)(ws + 2523136);               // 65536*40 f32
  _Float16* rW1h  = (_Float16*)(ws + 13008896);        // 128x64 frag-packed
  _Float16* rW2h  = (_Float16*)(ws + 13025280);        // 64x64 frag-packed
  _Float16* rW3h  = (_Float16*)(ws + 13033472);        // 64x2304 frag-packed
  _Float16* ecW1h = (_Float16*)(ws + 13328384);        // 128x64 frag-packed
  _Float16* ecW2h = (_Float16*)(ws + 13344768);        // 64x64 frag-packed

  float* fout = (float*)d_out;                 // f_new: 4096*80
  float* xout = fout + N_NODES * F_DIM;        // x_new: 4096*3

  // zero segment-sum accumulators (agg_s + agg_v + nupd, contiguous)
  zero_kernel<<<512, 256, 0, stream>>>(agg_s, (524288 + 786432 + 49152) / 4);
  prep_weights_kernel<<<672, 256, 0, stream>>>(rW1, rW2, rW3, ecW1, ecW2,
                                               rW1h, rW2h, rW3h, ecW1h, ecW2h);
  node_scalars_kernel<<<(N_NODES * 32 + 255) / 256, 256, 0, stream>>>(f, W_readout, ns);

  edge_msg_kernel<<<N_EDGES / 16, 256, 0, stream>>>(
      f, x, xA, xB, src, dst, bondA, bondB, ns,
      rW1h, rb1, rW2h, rb2, rW3h, rb3, efeat, agg_s, agg_v);

  node_update_kernel<<<(N_NODES + 255) / 256, 256, 0, stream>>>(
      f, agg_s, agg_v, W_si_s, W_si_v, W_pg_s, W_pg_v, W_readout, W_coord,
      cgW1, cgb1, cgW2, cgb2, cgW3, cgb3, fout, scal, dxv, g);

  edge_coord_kernel<<<N_EDGES / 16, 256, 0, stream>>>(
      x, src, dst, efeat, scal, ecW1h, ecb1, ecW2h, ecb2, ecW3, ecb3, nupd);

  coord_final_kernel<<<(N_NODES + 255) / 256, 256, 0, stream>>>(
      x, xA, xB, s, g, dxv, nupd, xout);
}
